// Abstract_Loss_71665824301698
// MI455X (gfx1250) — compile-verified, matrix operand order checked by reading, not run on hardware
//
#include <hip/hip_runtime.h>
#include <math.h>

#define BS 64
#define NF 256
#define NC 128
#define DD 768
#define BETA 0.1f
#define INFV 1000000000.0f

typedef __attribute__((ext_vector_type(16))) _Float16 v16h;
typedef __attribute__((ext_vector_type(8)))  float    v8f;

// workspace layout (in floats)
#define RNV_OFF   0                       // BS*NF inv-norms of video frame rows
#define RNT_OFF   (BS*NF)                 // BS*NC inv-norms of text cand rows
#define COST_OFF  (RNT_OFF + BS*NC)       // BS*NF*NC cost matrix
#define PART_OFF  (COST_OFF + BS*NF*NC)   // BS per-batch dtw values

// ---------------- Kernel 1: row inverse norms (wave per row) ----------------
__global__ void k_norms(const float* __restrict__ V, const float* __restrict__ T,
                        float* __restrict__ ws) {
  int w = threadIdx.x >> 5, lane = threadIdx.x & 31;
  int row = blockIdx.x * 8 + w;
  const float* p;
  float* out;
  if (row < BS * NF) { p = V + (size_t)row * DD; out = ws + RNV_OFF + row; }
  else { int r2 = row - BS * NF; p = T + (size_t)r2 * DD; out = ws + RNT_OFF + r2; }
  float s = 0.0f;
#pragma unroll
  for (int e = 0; e < DD / 32; e++) { float x = p[lane + 32 * e]; s += x * x; }
#pragma unroll
  for (int off = 16; off >= 1; off >>= 1) s += __shfl_xor(s, off, 32);
  if (lane == 0) *out = 1.0f / fmaxf(sqrtf(s), 1e-12f);
}

// convert 8 consecutive scaled f32 (two float4) into halves dst[off..off+7]
__device__ __forceinline__ void cvt8(const float4 x0, const float4 x1, float sc,
                                     v16h& dst, int off) {
  dst[off + 0] = (_Float16)(x0.x * sc); dst[off + 1] = (_Float16)(x0.y * sc);
  dst[off + 2] = (_Float16)(x0.z * sc); dst[off + 3] = (_Float16)(x0.w * sc);
  dst[off + 4] = (_Float16)(x1.x * sc); dst[off + 5] = (_Float16)(x1.y * sc);
  dst[off + 6] = (_Float16)(x1.z * sc); dst[off + 7] = (_Float16)(x1.w * sc);
}

// ---------------- Kernel 2: WMMA GEMM -> cost = 1 - sim ----------------
// grid (2, BS), 256 threads. wave owns m_tile = blockIdx.x*8 + wave, all 8 n-tiles.
__global__ void k_gemm(const float* __restrict__ V, const float* __restrict__ T,
                       float* __restrict__ ws) {
  const float* rnV = ws + RNV_OFF;
  const float* rnT = ws + RNT_OFF;
  float* cost = ws + COST_OFF;

  int b = blockIdx.y;
  int wave = threadIdx.x >> 5;
  int m_tile = blockIdx.x * 8 + wave;      // 0..15
  int lane = threadIdx.x & 31;
  int h = lane >> 4, ml = lane & 15;

  int m = m_tile * 16 + ml;
  const float* vrow = V + ((size_t)(b * NF + m)) * DD;
  float rv = rnV[b * NF + m];

  const float* trow[8];
  float rt[8];
#pragma unroll
  for (int j = 0; j < 8; j++) {
    int n = j * 16 + ml;
    trow[j] = T + ((size_t)(b * NC + n)) * DD;
    rt[j] = rnT[b * NC + n];
  }

  v8f acc[8];
#pragma unroll
  for (int j = 0; j < 8; j++)
#pragma unroll
    for (int r = 0; r < 8; r++) acc[j][r] = 0.0f;

  for (int k0 = 0; k0 < DD; k0 += 32) {
    int base = k0 + h * 8;                 // K = base..base+7 and base+16..base+23
    const float4* vp = (const float4*)(vrow + base);
    float4 a0 = vp[0], a1 = vp[1], a2 = vp[4], a3 = vp[5];
    v16h afrag;
    cvt8(a0, a1, rv, afrag, 0);
    cvt8(a2, a3, rv, afrag, 8);
#pragma unroll
    for (int j = 0; j < 8; j++) {
      const float4* tp = (const float4*)(trow[j] + base);
      float4 b0 = tp[0], b1 = tp[1], b2 = tp[4], b3 = tp[5];
      v16h bfrag;
      cvt8(b0, b1, rt[j], bfrag, 0);
      cvt8(b2, b3, rt[j], bfrag, 8);
      acc[j] = __builtin_amdgcn_wmma_f32_16x16x32_f16(
          false, afrag, false, bfrag, (short)0, acc[j], false, false);
    }
  }

#pragma unroll
  for (int j = 0; j < 8; j++) {
#pragma unroll
    for (int r = 0; r < 8; r++) {
      int mr = m_tile * 16 + h * 8 + r;
      int nc = j * 16 + ml;
      cost[((size_t)(b * NF + mr)) * NC + nc] = 1.0f - acc[j][r];
    }
  }
}

// ---------------- Kernel 3: soft-DTW wavefront DP (block per batch) ----------------
__global__ void k_dtw(const float* __restrict__ ws_c, float* __restrict__ ws) {
  const float* cost = ws_c + COST_OFF;
  float* partial = ws + PART_OFF;
  int b = blockIdx.x, t = threadIdx.x;
  int i = t + 1;                           // DP row index 1..256

  __shared__ float diag[3][NF + 4];        // 3 rotating anti-diagonals, idx 0..256
  diag[0][i] = INFV;                       // k=0 diagonal: R[0][0]=0, else INF
  diag[1][i] = INFV;                       // k=1 diagonal: all INF
  if (t == 0) { diag[0][0] = 0.0f; diag[1][0] = INFV; }

  // each thread streams cost row (i-1) sequentially in j, float4 double-buffer
  const float4* row4 = (const float4*)(cost + ((size_t)(b * NF + (i - 1))) * NC);
  float4 cur4 = row4[0];
  float4 nxt4 = row4[1];
  __syncthreads();

  for (int k = 2; k <= NF + NC; k++) {
    float* cur = diag[k % 3];              // will hold diagonal k (overwrites k-3)
    const float* p1 = diag[(k + 2) % 3];   // diagonal k-1
    const float* p2 = diag[(k + 1) % 3];   // diagonal k-2
    int j = k - i - 1;
    float nv = INFV;
    if (j >= 0 && j < NC) {
      int q = j >> 2, rm = j & 3;
      if (rm == 0 && q > 0) {
        cur4 = nxt4;
        if (q < (NC / 4) - 1) nxt4 = row4[q + 1];   // prefetch 4 steps ahead
      }
      float d = (rm == 0) ? cur4.x : (rm == 1) ? cur4.y : (rm == 2) ? cur4.z : cur4.w;
      float a  = p1[i - 1];                // R[i-1][j]
      float bb = p1[i];                    // R[i][j-1]
      float cc = p2[i - 1];                // R[i-1][j-1]
      float mn = fminf(a, fminf(bb, cc));
      float s = __expf((mn - a) * (1.0f / BETA)) +
                __expf((mn - bb) * (1.0f / BETA)) +
                __expf((mn - cc) * (1.0f / BETA));
      nv = d + mn - BETA * __logf(s);
    }
    cur[i] = nv;
    if (t == 0) cur[0] = INFV;
    __syncthreads();
  }
  // final diagonal k = NF+NC = 384, slot 384%3 == 0, result at index NF
  if (t == 255) partial[b] = diag[(NF + NC) % 3][NF] * (1.0f / (float)(NF + NC));
}

// ---------------- Kernel 4: deterministic final reduction ----------------
__global__ void k_reduce(const float* __restrict__ ws, float* __restrict__ out) {
  __shared__ float red[BS];
  int t = threadIdx.x;
  red[t] = ws[PART_OFF + t];
  __syncthreads();
  if (t == 0) {
    float s = 0.0f;
    for (int q = 0; q < BS; q++) s += red[q];
    out[0] = s * (1.0f / (float)BS);      // DTW_RATIO = 1
  }
}

extern "C" void kernel_launch(void* const* d_in, const int* in_sizes, int n_in,
                              void* d_out, int out_size, void* d_ws, size_t ws_size,
                              hipStream_t stream) {
  (void)in_sizes; (void)n_in; (void)out_size; (void)ws_size;
  const float* V = (const float*)d_in[1];   // video_embd_frame (64,256,768)
  const float* T = (const float*)d_in[3];   // text_embd_cand  (64,128,768)
  float* ws = (float*)d_ws;
  float* out = (float*)d_out;

  k_norms <<<dim3((BS * (NF + NC)) / 8), 256, 0, stream>>>(V, T, ws);
  k_gemm  <<<dim3(2, BS),               256, 0, stream>>>(V, T, ws);
  k_dtw   <<<dim3(BS),                  256, 0, stream>>>(ws, ws);
  k_reduce<<<1, BS, 0, stream>>>(ws, out);
}
